// CustomCausalAttention_59837484368477
// MI455X (gfx1250) — matrix operand order repaired and checked by reading, not compiled, not run on hardware
//
#include <hip/hip_runtime.h>

// ---------------- CDNA5 (gfx1250, wave32) types & helpers ----------------
typedef __bf16 bf16_t;
typedef __attribute__((ext_vector_type(16))) __bf16 v16bf;
typedef __attribute__((ext_vector_type(8)))  __bf16 v8bf;
typedef __attribute__((ext_vector_type(4)))  float  v4f;
typedef __attribute__((ext_vector_type(8)))  float  v8f;

#define WMMA_BF16(a, b, c) \
  __builtin_amdgcn_wmma_f32_16x16x32_bf16(false, (a), false, (b), (short)0, (c), false, false)

union AFrag { v16bf v; v8bf h[2]; };

constexpr int BB = 4;
constexpr int TT = 2048;
constexpr int DD = 1024;
constexpr float SCL = 0.03125f; // 1/sqrt(1024)

// ---------------- Kernel 0a: convert activations x -> bf16 ----------------
// xb[b][t][d] = (bf16) x[b][t][d]; 4 elements/thread, fully coalesced.
__global__ __launch_bounds__(256)
void xb_kernel(const float* __restrict__ x, bf16_t* __restrict__ xb) {
  size_t i = ((size_t)blockIdx.x * 256 + threadIdx.x) * 4;
  v4f f = *(const v4f*)(x + i);
  v8bf* o = (v8bf*)(xb + i);     // store 4 bf16 = 8B; use half of a v8bf slot
  bf16_t tmp[4] = {(bf16_t)f.x, (bf16_t)f.y, (bf16_t)f.z, (bf16_t)f.w};
  *(uint2*)(xb + i) = *(const uint2*)tmp;
  (void)o;
}

// ---------------- Kernel 0b: transpose-convert weights to bf16 ------------
// wt[e][d] = (bf16) w[d][e]   (so WMMA B-fragments are contiguous per lane)
__global__ __launch_bounds__(256)
void wt_kernel(const float* __restrict__ wq, const float* __restrict__ wk,
               const float* __restrict__ wv,
               bf16_t* __restrict__ tq, bf16_t* __restrict__ tk, bf16_t* __restrict__ tv) {
  int z = blockIdx.z;
  const float* w = (z == 0) ? wq : (z == 1) ? wk : wv;
  bf16_t*      o = (z == 0) ? tq : (z == 1) ? tk : tv;
  int idx = blockIdx.x * 256 + threadIdx.x;   // over DD*DD
  int e = idx & (DD - 1);
  int d = idx >> 10;
  o[(size_t)e * DD + d] = (bf16_t)w[idx];
}

// ---------------- Kernel 1: Q/K/V projections via bf16 WMMA ---------------
// z=0: Qb[b][t][e], z=1: Kb[b][t][e], z=2: Vt[b][e][t] (transposed store)
__global__ __launch_bounds__(256)
void proj_kernel(const bf16_t* __restrict__ xb,
                 const bf16_t* __restrict__ wtq, const bf16_t* __restrict__ wtk,
                 const bf16_t* __restrict__ wtv,
                 bf16_t* __restrict__ Qb, bf16_t* __restrict__ Kb, bf16_t* __restrict__ Vt) {
  const int z = blockIdx.z;
  const bf16_t* wt = (z == 0) ? wtq : (z == 1) ? wtk : wtv;

  const int tile = blockIdx.x;            // 0 .. BB*TT/16-1
  const int b  = tile / (TT / 16);
  const int t0 = (tile % (TT / 16)) * 16;
  const int wave = threadIdx.x >> 5;
  const int lane = threadIdx.x & 31;
  const int l15  = lane & 15;
  const int hi   = lane >> 4;
  const int eb   = blockIdx.y * 128 + wave * 16;   // this wave's 16-wide E tile

  const bf16_t* xrow = xb + ((size_t)(b * TT + t0 + l15)) * DD;
  const bf16_t* wrow = wt + (size_t)(eb + l15) * DD;

  v8f acc = {};
  for (int k0 = 0; k0 < DD; k0 += 32) {
    // A fragment: row (t0+l15), K {k0+hi*16 ..+16} contiguous
    AFrag f;
    f.h[0] = *(const v8bf*)(xrow + k0 + hi * 16);
    f.h[1] = *(const v8bf*)(xrow + k0 + hi * 16 + 8);
    // B fragment: column e = eb+l15, K-contiguous 16 bf16 (32B)
    v16bf bf = *(const v16bf*)(wrow + k0 + hi * 16);
    acc = WMMA_BF16(f.v, bf, acc);
  }

  // C layout: acc[i] -> (M = i + 8*hi, N = l15)
  if (z == 2) {
#pragma unroll
    for (int i = 0; i < 8; ++i) {
      int t = t0 + i + 8 * hi;
      int e = eb + l15;
      Vt[((size_t)b * DD + e) * TT + t] = (bf16_t)acc[i];
    }
  } else {
    bf16_t* out = (z == 0) ? Qb : Kb;
#pragma unroll
    for (int i = 0; i < 8; ++i) {
      int t = t0 + i + 8 * hi;
      out[((size_t)(b * TT + t)) * DD + eb + l15] = (bf16_t)acc[i];
    }
  }
}

// ---------------- Kernel 2: causal flash attention -----------------------
// One block (8 waves) per (batch, 16-query tile). Wave w owns d-slice
// [w*128, w*128+128) for both the QK^T contraction and the O columns.
__global__ __launch_bounds__(256)
void attn_kernel(const bf16_t* __restrict__ Qb, const bf16_t* __restrict__ Kb,
                 const bf16_t* __restrict__ Vt, float* __restrict__ out) {
  __shared__ __align__(16) float  S_lds[8][16][32];   // per-wave partial S slabs
  __shared__ __align__(16) bf16_t P_lds[16][32];      // probs tile (bf16, row-major)
  __shared__ __align__(16) float  m_s[16], l_s[16], sc_s[16];

  const int blk = blockIdx.x;                 // 0 .. BB*TT/16-1
  const int b  = blk / (TT / 16);
  const int q0 = (blk % (TT / 16)) * 16;
  const int tid  = threadIdx.x;
  const int wave = tid >> 5;
  const int lane = tid & 31;
  const int l15  = lane & 15;
  const int hi   = lane >> 4;
  const int d0   = wave * 128;

  if (tid < 16) { m_s[tid] = -__builtin_inff(); l_s[tid] = 0.0f; }

  // Preload this wave's Q A-fragments (16 rows x 128 d) into registers.
  const bf16_t* qrow = Qb + ((size_t)(b * TT + q0 + l15)) * DD + d0;
  v16bf aq[4];
#pragma unroll
  for (int c = 0; c < 4; ++c) {
    AFrag f;
    f.h[0] = *(const v8bf*)(qrow + c * 32 + hi * 16);
    f.h[1] = *(const v8bf*)(qrow + c * 32 + hi * 16 + 8);
    aq[c] = f.v;
  }

  v8f acc[8] = {};  // O accumulators: 8 N-tiles of 16 over this wave's 128 cols

  const bf16_t* kbase = Kb + (size_t)b * TT * DD;
  const bf16_t* vbase = Vt + (size_t)b * DD * TT;
  const int ntiles = (q0 + 15) / 32 + 1;

  __syncthreads();

  for (int it = 0; it < ntiles; ++it) {
    const int kv0 = it * 32;

    // ---- partial S = Q * K^T over this wave's d-slice ----
#pragma unroll
    for (int nh = 0; nh < 2; ++nh) {
      const bf16_t* krow = kbase + (size_t)(kv0 + nh * 16 + l15) * DD + d0;
      if (it + 1 < ntiles)  // prefetch next K tile rows -> global_prefetch_b8
        __builtin_prefetch(krow + 32 * DD, 0, 0);
      v8f cs = {};
#pragma unroll
      for (int c = 0; c < 4; ++c) {
        v16bf bf = *(const v16bf*)(krow + c * 32 + hi * 16);
        cs = WMMA_BF16(aq[c], bf, cs);
      }
#pragma unroll
      for (int i = 0; i < 8; ++i)
        S_lds[wave][i + 8 * hi][nh * 16 + l15] = cs[i];
    }
    __syncthreads();

    // ---- online softmax: thread handles row r = tid/16, cols {2c, 2c+1} ----
    {
      const int r  = tid >> 4;
      const int cc = (tid & 15) * 2;
      float s0 = 0.0f, s1 = 0.0f;
#pragma unroll
      for (int w = 0; w < 8; ++w) { s0 += S_lds[w][r][cc]; s1 += S_lds[w][r][cc + 1]; }
      s0 *= SCL; s1 *= SCL;
      const int qg = q0 + r;
      if (kv0 + cc     > qg) s0 = -__builtin_inff();
      if (kv0 + cc + 1 > qg) s1 = -__builtin_inff();
      float mx = fmaxf(s0, s1);
#pragma unroll
      for (int m = 8; m >= 1; m >>= 1) mx = fmaxf(mx, __shfl_xor(mx, m, 16));
      const float m_old = m_s[r], l_old = l_s[r];
      const float m_new = fmaxf(m_old, mx);
      const float p0 = __expf(s0 - m_new);
      const float p1 = __expf(s1 - m_new);
      float lt = p0 + p1;
#pragma unroll
      for (int m = 8; m >= 1; m >>= 1) lt += __shfl_xor(lt, m, 16);
      const float sc = __expf(m_old - m_new);
      P_lds[r][cc]     = (bf16_t)p0;
      P_lds[r][cc + 1] = (bf16_t)p1;
      if ((tid & 15) == 0) { m_s[r] = m_new; l_s[r] = l_old * sc + lt; sc_s[r] = sc; }
    }
    __syncthreads();

    // ---- rescale O accumulators by exp(m_old - m_new) per row ----
    float sc[8];
#pragma unroll
    for (int i = 0; i < 8; ++i) sc[i] = sc_s[i + 8 * hi];
#pragma unroll
    for (int j = 0; j < 8; ++j)
#pragma unroll
      for (int i = 0; i < 8; ++i) acc[j][i] *= sc[i];

    // ---- O += P * V : A from P_lds (LDS), B from Vt (global, contiguous) ----
    AFrag pf;
    pf.h[0] = *(const v8bf*)&P_lds[l15][hi * 16];
    pf.h[1] = *(const v8bf*)&P_lds[l15][hi * 16 + 8];
    const v16bf ap = pf.v;
#pragma unroll
    for (int j = 0; j < 8; ++j) {
      const int nb = d0 + j * 16;
      const bf16_t* vrow = vbase + (size_t)(nb + l15) * TT + kv0 + hi * 16;
      if (it + 1 < ntiles) __builtin_prefetch(vrow + 32, 0, 0);
      v16bf bf = *(const v16bf*)vrow;
      acc[j] = WMMA_BF16(ap, bf, acc[j]);
    }
    __syncthreads();  // protect S_lds / P_lds reuse next iteration
  }

  // ---- epilogue: divide by l, write fp32 output ----
  float invl[8];
#pragma unroll
  for (int i = 0; i < 8; ++i) invl[i] = 1.0f / l_s[i + 8 * hi];
#pragma unroll
  for (int j = 0; j < 8; ++j) {
    const int nb = d0 + j * 16;
#pragma unroll
    for (int i = 0; i < 8; ++i) {
      const int t = q0 + i + 8 * hi;
      out[((size_t)(b * TT + t)) * DD + nb + l15] = acc[j][i] * invl[i];
    }
  }
}

// ---------------- launch -------------------------------------------------
extern "C" void kernel_launch(void* const* d_in, const int* in_sizes, int n_in,
                              void* d_out, int out_size, void* d_ws, size_t ws_size,
                              hipStream_t stream) {
  const float* x  = (const float*)d_in[0];
  const float* wq = (const float*)d_in[1];
  const float* wk = (const float*)d_in[2];
  const float* wv = (const float*)d_in[3];
  float* out = (float*)d_out;

  // Workspace layout (bf16): xb | Qb | Kb | Vt | WqT | WkT | WvT  (70 MB)
  bf16_t* xbp = (bf16_t*)d_ws;
  bf16_t* Qb  = xbp + (size_t)BB * TT * DD;
  bf16_t* Kb  = Qb  + (size_t)BB * TT * DD;
  bf16_t* Vt  = Kb  + (size_t)BB * TT * DD;
  bf16_t* wtq = Vt  + (size_t)BB * TT * DD;
  bf16_t* wtk = wtq + (size_t)DD * DD;
  bf16_t* wtv = wtk + (size_t)DD * DD;

  xb_kernel<<<dim3((size_t)BB * TT * DD / (256 * 4)), 256, 0, stream>>>(x, xbp);

  wt_kernel<<<dim3(DD * DD / 256, 1, 3), 256, 0, stream>>>(wq, wk, wv, wtq, wtk, wtv);

  proj_kernel<<<dim3(BB * TT / 16, DD / 128, 3), 256, 0, stream>>>(
      xbp, wtq, wtk, wtv, Qb, Kb, Vt);

  attn_kernel<<<dim3(BB * TT / 16, 1, 1), 256, 0, stream>>>(Qb, Kb, Vt, out);
}
